// MultiHeadConvAttn_58514634441416
// MI455X (gfx1250) — matrix operand order repaired and checked by reading, not compiled, not run on hardware
//
#include <hip/hip_runtime.h>
#include <hip/hip_bf16.h>
#include <math.h>

#define BATCH 2
#define SEQ   2048
#define DIM   1024
#define NHEAD 16
#define DHEAD 64
#define COUTC 512
#define NTOK  (BATCH*SEQ)

typedef __attribute__((ext_vector_type(16))) _Float16 v16h;
typedef __attribute__((ext_vector_type(8)))  _Float16 v8h;
typedef __attribute__((ext_vector_type(8)))  float    v8f;

// ---- WMMA operand loaders (wave32, v_wmma_f32_16x16x32_f16) ----
// A operand: lane holds row M=lane&15; K elems at [khalf*8..+7] and [16+khalf*8..+7]
// B operand: lane holds col N=lane&15; K elems at [khalf*8..+7] and [16+khalf*8..+7]
// Caller passes p already offset by row*lda + khalf*8 (A) or col*ldb + khalf*8 (B).

__device__ __forceinline__ v16h load_a_f16(const _Float16* p) {
    v16h r;
    v8h lo = *(const v8h*)p;
    v8h hi = *(const v8h*)(p + 16);
#pragma unroll
    for (int i = 0; i < 8; ++i) { r[i] = lo[i]; r[8 + i] = hi[i]; }
    return r;
}

__device__ __forceinline__ v16h load_b_f16(const _Float16* p) {
    v16h r;
    v8h lo = *(const v8h*)p;
    v8h hi = *(const v8h*)(p + 16);
#pragma unroll
    for (int i = 0; i < 8; ++i) { r[i] = lo[i]; r[8 + i] = hi[i]; }
    return r;
}

__device__ __forceinline__ v8f wmma16(v16h a, v16h b, v8f c) {
    return __builtin_amdgcn_wmma_f32_16x16x32_f16(false, a, false, b, (short)0, c,
                                                  false, false);
}

// ---- Kernel 0a: fp32 -> f16, 4 elems/thread ----
__global__ void cvt_kernel(const float* __restrict__ in, _Float16* __restrict__ out,
                           int n4) {
    int t = blockIdx.x * blockDim.x + threadIdx.x;
    if (t >= n4) return;
    float4 v = *(const float4*)(in + (size_t)t * 4);
    _Float16* o = out + (size_t)t * 4;
    o[0] = (_Float16)v.x; o[1] = (_Float16)v.y;
    o[2] = (_Float16)v.z; o[3] = (_Float16)v.w;
}

// ---- Kernel 0b: conv weight repack OIH [co][ci][t] -> f16 [co][t][ci] ----
__global__ void wrepack_kernel(const float* __restrict__ w, _Float16* __restrict__ wt,
                               int ksz) {
    int idx = blockIdx.x * blockDim.x + threadIdx.x;  // output-major index
    int total = COUTC * DIM * ksz;
    if (idx >= total) return;
    int ci = idx % DIM;
    int t  = (idx / DIM) % ksz;
    int co = idx / (DIM * ksz);
    wt[idx] = (_Float16)w[((size_t)co * DIM + ci) * ksz + t];
}

// ---- Kernel 1: C_f16 = A_f16 @ W_f16^T, 32(M)x64(N) tile/wave; opt. transposed V ----
__global__ void proj_kernel(const _Float16* __restrict__ A,
                            const _Float16* __restrict__ W,
                            _Float16* __restrict__ out, int Kdim, int Ndim,
                            int transpose_v) {
    const int lane  = threadIdx.x & 31;
    const int row   = lane & 15;
    const int khalf = lane >> 4;
    const int col   = lane & 15;
    const int m_base = blockIdx.x * 32;
    const int n_base = blockIdx.y * 64;

    const _Float16* arow0 = A + (size_t)(m_base + row) * Kdim + khalf * 8;
    const _Float16* arow1 = arow0 + (size_t)16 * Kdim;
    const _Float16* brow  = W + (size_t)(n_base + row) * Kdim + khalf * 8;

    v8f c[2][4] = {};
    for (int k = 0; k < Kdim; k += 32) {
        if (k + 64 < Kdim) {  // gfx1250 global_prefetch_b8
            __builtin_prefetch(arow0 + k + 64, 0, 0);
            __builtin_prefetch(brow + k + 64, 0, 0);
        }
        v16h a0 = load_a_f16(arow0 + k);
        v16h a1 = load_a_f16(arow1 + k);
#pragma unroll
        for (int nt = 0; nt < 4; ++nt) {
            v16h b = load_b_f16(brow + (size_t)nt * 16 * Kdim + k);
            c[0][nt] = wmma16(a0, b, c[0][nt]);
            c[1][nt] = wmma16(a1, b, c[1][nt]);
        }
    }

    if (!transpose_v) {
#pragma unroll
        for (int mt = 0; mt < 2; ++mt)
#pragma unroll
            for (int nt = 0; nt < 4; ++nt)
#pragma unroll
                for (int r = 0; r < 8; ++r) {
                    int m = m_base + mt * 16 + r + 8 * khalf;
                    out[(size_t)m * Ndim + n_base + nt * 16 + col] =
                        (_Float16)c[mt][nt][r];
                }
    } else {  // V stored [B,H,DH,S] so P@V B-operand reads are contiguous in k
#pragma unroll
        for (int nt = 0; nt < 4; ++nt) {
            int n = n_base + nt * 16 + col;
            int h = n >> 6, dh = n & (DHEAD - 1);
#pragma unroll
            for (int mt = 0; mt < 2; ++mt)
#pragma unroll
                for (int r = 0; r < 8; ++r) {
                    int m = m_base + mt * 16 + r + 8 * khalf;
                    int bb = m >> 11, s = m & (SEQ - 1);
                    out[(((size_t)(bb * NHEAD + h)) * DHEAD + dh) * SEQ + s] =
                        (_Float16)c[mt][nt][r];
                }
        }
    }
}

// ---- Kernel 2: conv branch as implicit GEMM (K order t*D+ci), 64(M)x16(N)/wave ----
__global__ void conv_kernel(const _Float16* __restrict__ k16,
                            const _Float16* __restrict__ wt,  // [co][t][ci] f16
                            const float* __restrict__ bias,
                            _Float16* __restrict__ out,       // [B,S,D] f16
                            int ksz, int co_off) {
    const int lane  = threadIdx.x & 31;
    const int row   = lane & 15;
    const int khalf = lane >> 4;
    const int m_base  = blockIdx.x * 64;      // token tile (never straddles batch)
    const int co_base = blockIdx.y * 16;
    const int bb = m_base / SEQ;
    const int s_base = m_base & (SEQ - 1);
    const int pad = ksz >> 1;
    const int co = co_base + (lane & 15);

    const _Float16* wb = wt + ((size_t)co * ksz) * DIM + khalf * 8;

    v8f c[4] = {};
    for (int t = 0; t < ksz; ++t) {
        const _Float16* wrow = wb + (size_t)t * DIM;
        for (int ci = 0; ci < DIM; ci += 32) {
            v16h bm = load_b_f16(wrow + ci);
#pragma unroll
            for (int mt = 0; mt < 4; ++mt) {
                int s_src = s_base + mt * 16 + row + t - pad;
                bool valid = (s_src >= 0) && (s_src < SEQ);
                v16h a = {};
                if (valid)
                    a = load_a_f16(k16 + ((size_t)bb * SEQ + s_src) * DIM + ci +
                                   khalf * 8);
                c[mt] = wmma16(a, bm, c[mt]);
            }
        }
    }

    float bv = bias[co];
#pragma unroll
    for (int mt = 0; mt < 4; ++mt)
#pragma unroll
        for (int r = 0; r < 8; ++r) {
            int m = m_base + mt * 16 + r + 8 * khalf;
            out[(size_t)m * DIM + co_off + co] = (_Float16)(c[mt][r] + bv);
        }
}

// ---- Kernel 3: k_blend = (1-g_h)*k_proj + g_h*k_conv, f16 ----
__global__ void blend_kernel(const _Float16* __restrict__ kproj,
                             const _Float16* __restrict__ kconv,
                             const float* __restrict__ gate,
                             _Float16* __restrict__ kblend) {
    size_t idx = (size_t)blockIdx.x * blockDim.x + threadIdx.x;
    if (idx >= (size_t)NTOK * DIM) return;
    int d = (int)(idx & (DIM - 1));
    int h = d >> 6;
    float g = 1.f / (1.f + __expf(-gate[h]));
    float v = (1.f - g) * (float)kproj[idx] + g * (float)kconv[idx];
    kblend[idx] = (_Float16)v;
}

// LDS row strides (halfs / floats) padded for conflict-free 64-bank access
#define KBUF_LD 72   // 144B rows: bank stride 36 -> 16 distinct 4-bank groups
#define VBUF_LD 40   // 80B rows:  bank stride 20 -> distinct
#define PBUF_LD 40
#define SBUF_LD 33

// ---- Kernel 4: flash attention; async double-buffered K/V tile staging in LDS ----
__global__ void attn_kernel(const _Float16* __restrict__ qproj,
                            const _Float16* __restrict__ kblend,
                            const _Float16* __restrict__ vT,
                            const int* __restrict__ mask,
                            _Float16* __restrict__ attn_out) {
    const int lane  = threadIdx.x & 31;
    const int khalf = lane >> 4;
    const int col   = lane & 15;
    const int q_base = blockIdx.x * 16;
    const int bh = blockIdx.y;
    const int bb = bh / NHEAD, h = bh % NHEAD;

    __shared__ _Float16 kbuf[2][32 * KBUF_LD];   // 32 key rows x 64 halfs (padded)
    __shared__ _Float16 vbuf[2][64 * VBUF_LD];   // 64 d rows x 32 halfs (padded)
    __shared__ float    s_lds[16 * SBUF_LD];
    __shared__ _Float16 p_lds[16 * PBUF_LD];
    __shared__ float    alpha_lds[16];
    __shared__ float    linv_lds[16];

    const size_t tokq = (size_t)bb * SEQ + q_base;
    const _Float16* qrow = qproj + (tokq + col) * DIM + h * DHEAD + khalf * 8;
    v16h aq0 = load_a_f16(qrow);
    v16h aq1 = load_a_f16(qrow + 32);

    // uniform global bases for async staging (saddr in SGPRs, per-lane VGPR offsets)
    const unsigned long long kg =
        (unsigned long long)(uintptr_t)(kblend + ((size_t)bb * SEQ) * DIM + h * DHEAD);
    const unsigned long long vg =
        (unsigned long long)(uintptr_t)(vT + ((size_t)(bb * NHEAD + h) * DHEAD) * SEQ);
    const uint32_t kbuf_a0 = (uint32_t)(uintptr_t)&kbuf[0][0];
    const uint32_t kbuf_a1 = (uint32_t)(uintptr_t)&kbuf[1][0];
    const uint32_t vbuf_a0 = (uint32_t)(uintptr_t)&vbuf[0][0];
    const uint32_t vbuf_a1 = (uint32_t)(uintptr_t)&vbuf[1][0];

    // issue one (K,V) tile pair: 16x global_load_async_to_lds_b128 (ASYNCcnt +16)
    auto issue_tiles = [&](int kc, int buf) {
        uint32_t kl = (buf ? kbuf_a1 : kbuf_a0) + (uint32_t)(lane * (KBUF_LD * 2));
        uint32_t kv = (uint32_t)((kc + lane) * DIM * 2);
#pragma unroll
        for (int c2 = 0; c2 < 8; ++c2)
            asm volatile("global_load_async_to_lds_b128 %0, %1, %2"
                         :: "v"(kl + 16 * c2), "v"(kv + 16 * c2), "s"(kg)
                         : "memory");
#pragma unroll
        for (int rg = 0; rg < 2; ++rg) {
            int d = lane + rg * 32;
            uint32_t vl = (buf ? vbuf_a1 : vbuf_a0) + (uint32_t)(d * (VBUF_LD * 2));
            uint32_t vv = (uint32_t)((d * SEQ + kc) * 2);
#pragma unroll
            for (int c2 = 0; c2 < 4; ++c2)
                asm volatile("global_load_async_to_lds_b128 %0, %1, %2"
                             :: "v"(vl + 16 * c2), "v"(vv + 16 * c2), "s"(vg)
                             : "memory");
        }
    };

    v8f acc0 = {}, acc1 = {}, acc2 = {}, acc3 = {};
    float m_run = -3.0e38f, l_run = 0.f;
    const float scale = 0.125f;  // 1/sqrt(64)

    issue_tiles(0, 0);

    for (int kc = 0; kc < SEQ; kc += 32) {
        const int cur = (kc >> 5) & 1;
        if (kc + 32 < SEQ) {
            issue_tiles(kc + 32, cur ^ 1);
            asm volatile("s_wait_asynccnt 0x10" ::: "memory");  // current tile done
        } else {
            asm volatile("s_wait_asynccnt 0x0" ::: "memory");
        }
        const _Float16* kb = &kbuf[cur][0];
        const _Float16* vb = &vbuf[cur][0];

        // ---- scores from LDS-staged K tile ----
        v8f s0 = {}, s1 = {};
        {
            const _Float16* kp0 = kb + col * KBUF_LD + khalf * 8;
            const _Float16* kp1 = kp0 + 16 * KBUF_LD;
            s0 = wmma16(aq0, load_b_f16(kp0), s0);
            s0 = wmma16(aq1, load_b_f16(kp0 + 32), s0);
            s1 = wmma16(aq0, load_b_f16(kp1), s1);
            s1 = wmma16(aq1, load_b_f16(kp1 + 32), s1);
        }
        // ---- scale + mask -> LDS [row][col] ----
#pragma unroll
        for (int r = 0; r < 8; ++r) {
            int rr = r + 8 * khalf;
            size_t mrow = (tokq + rr) * SEQ;
            float v0 = s0[r] * scale;
            float v1 = s1[r] * scale;
            if (mask[mrow + kc + col] == 0) v0 = -1e9f;
            if (mask[mrow + kc + 16 + col] == 0) v1 = -1e9f;
            s_lds[rr * SBUF_LD + col] = v0;
            s_lds[rr * SBUF_LD + 16 + col] = v1;
        }
        __syncthreads();
        // ---- online softmax row stats (lanes 0..15 own a row) ----
        if (lane < 16) {
            float mx = m_run;
#pragma unroll
            for (int j = 0; j < 32; ++j) mx = fmaxf(mx, s_lds[lane * SBUF_LD + j]);
            float alpha = __expf(m_run - mx);
            float sum = 0.f;
#pragma unroll
            for (int j = 0; j < 32; ++j) {
                float p = __expf(s_lds[lane * SBUF_LD + j] - mx);
                p_lds[lane * PBUF_LD + j] = (_Float16)p;
                sum += p;
            }
            l_run = l_run * alpha + sum;
            m_run = mx;
            alpha_lds[lane] = alpha;
        }
        __syncthreads();
#pragma unroll
        for (int r = 0; r < 8; ++r) {
            float al = alpha_lds[r + 8 * khalf];
            acc0[r] *= al; acc1[r] *= al; acc2[r] *= al; acc3[r] *= al;
        }
        // ---- P @ V from LDS-staged V tile ----
        v16h pa = load_a_f16(p_lds + col * PBUF_LD + khalf * 8);
        const _Float16* vp = vb + (size_t)col * VBUF_LD + khalf * 8;
        acc0 = wmma16(pa, load_b_f16(vp), acc0); vp += 16 * VBUF_LD;
        acc1 = wmma16(pa, load_b_f16(vp), acc1); vp += 16 * VBUF_LD;
        acc2 = wmma16(pa, load_b_f16(vp), acc2); vp += 16 * VBUF_LD;
        acc3 = wmma16(pa, load_b_f16(vp), acc3);
        __syncthreads();
    }

    if (lane < 16) linv_lds[lane] = 1.f / l_run;
    __syncthreads();

#pragma unroll
    for (int r = 0; r < 8; ++r) {
        int rr = r + 8 * khalf;
        float li = linv_lds[rr];
        size_t orow = (tokq + rr) * DIM + h * DHEAD;
        attn_out[orow + col]      = (_Float16)(acc0[r] * li);
        attn_out[orow + 16 + col] = (_Float16)(acc1[r] * li);
        attn_out[orow + 32 + col] = (_Float16)(acc2[r] * li);
        attn_out[orow + 48 + col] = (_Float16)(acc3[r] * li);
    }
}

// ---- Kernel 5: out = attn_f16 @ Wo16^T + bo, 32x64 tile/wave, fp32 out ----
__global__ void outproj_kernel(const _Float16* __restrict__ A,
                               const _Float16* __restrict__ Wo16,
                               const float* __restrict__ bo,
                               float* __restrict__ out) {
    const int lane  = threadIdx.x & 31;
    const int row   = lane & 15;
    const int khalf = lane >> 4;
    const int col   = lane & 15;
    const int m_base = blockIdx.x * 32;
    const int n_base = blockIdx.y * 64;

    const _Float16* arow0 = A + (size_t)(m_base + row) * DIM + khalf * 8;
    const _Float16* arow1 = arow0 + (size_t)16 * DIM;
    const _Float16* brow = Wo16 + (size_t)(n_base + row) * DIM + khalf * 8;

    v8f c[2][4] = {};
    for (int k = 0; k < DIM; k += 32) {
        if (k + 64 < DIM) {
            __builtin_prefetch(arow0 + k + 64, 0, 0);
            __builtin_prefetch(brow + k + 64, 0, 0);
        }
        v16h a0 = load_a_f16(arow0 + k);
        v16h a1 = load_a_f16(arow1 + k);
#pragma unroll
        for (int nt = 0; nt < 4; ++nt) {
            v16h b = load_b_f16(brow + (size_t)nt * 16 * DIM + k);
            c[0][nt] = wmma16(a0, b, c[0][nt]);
            c[1][nt] = wmma16(a1, b, c[1][nt]);
        }
    }
#pragma unroll
    for (int nt = 0; nt < 4; ++nt) {
        float bv = bo[n_base + nt * 16 + col];
#pragma unroll
        for (int mt = 0; mt < 2; ++mt)
#pragma unroll
            for (int r = 0; r < 8; ++r) {
                int m = m_base + mt * 16 + r + 8 * khalf;
                out[(size_t)m * DIM + n_base + nt * 16 + col] = c[mt][nt][r] + bv;
            }
    }
}

extern "C" void kernel_launch(void* const* d_in, const int* in_sizes, int n_in,
                              void* d_out, int out_size, void* d_ws, size_t ws_size,
                              hipStream_t stream) {
    const float* q    = (const float*)d_in[0];
    const float* k    = (const float*)d_in[1];
    const float* v    = (const float*)d_in[2];
    const int*   mask = (const int*)d_in[3];
    const float* Wq   = (const float*)d_in[4];
    const float* Wk   = (const float*)d_in[5];
    const float* Wv   = (const float*)d_in[6];
    const float* Wo   = (const float*)d_in[7];
    const float* bo   = (const float*)d_in[8];
    const float* cw0  = (const float*)d_in[9];
    const float* cb0  = (const float*)d_in[10];
    const float* cw1  = (const float*)d_in[11];
    const float* cb1  = (const float*)d_in[12];
    const float* gate = (const float*)d_in[13];

    char* p = (char*)d_ws;
    const size_t act_sz = (size_t)NTOK * DIM * sizeof(_Float16);   // 8 MiB
    const size_t w_sz   = (size_t)DIM * DIM * sizeof(_Float16);    // 2 MiB
    _Float16* q16    = (_Float16*)p; p += act_sz;
    _Float16* k16    = (_Float16*)p; p += act_sz;
    _Float16* v16    = (_Float16*)p; p += act_sz;
    _Float16* Wq16   = (_Float16*)p; p += w_sz;
    _Float16* Wk16   = (_Float16*)p; p += w_sz;
    _Float16* Wv16   = (_Float16*)p; p += w_sz;
    _Float16* Wo16   = (_Float16*)p; p += w_sz;
    _Float16* wt0    = (_Float16*)p; p += (size_t)COUTC * DIM * 3 * sizeof(_Float16);
    _Float16* wt1    = (_Float16*)p; p += (size_t)COUTC * DIM * 5 * sizeof(_Float16);
    _Float16* qproj  = (_Float16*)p; p += act_sz;
    _Float16* kproj  = (_Float16*)p; p += act_sz;
    _Float16* vT     = (_Float16*)p; p += act_sz;   // [B,H,DH,S]
    _Float16* kconv  = (_Float16*)p; p += act_sz;   // [B,S,D] channel-last
    _Float16* kblend = (_Float16*)p; p += act_sz;
    _Float16* attnO  = (_Float16*)p; p += act_sz;

    // 0) one-time fp32 -> f16 conversions (pure streaming, 23.3 TB/s bound)
    const int nact4 = NTOK * DIM / 4, nw4 = DIM * DIM / 4;
    cvt_kernel<<<(nact4 + 255) / 256, 256, 0, stream>>>(q, q16, nact4);
    cvt_kernel<<<(nact4 + 255) / 256, 256, 0, stream>>>(k, k16, nact4);
    cvt_kernel<<<(nact4 + 255) / 256, 256, 0, stream>>>(v, v16, nact4);
    cvt_kernel<<<(nw4 + 255) / 256, 256, 0, stream>>>(Wq, Wq16, nw4);
    cvt_kernel<<<(nw4 + 255) / 256, 256, 0, stream>>>(Wk, Wk16, nw4);
    cvt_kernel<<<(nw4 + 255) / 256, 256, 0, stream>>>(Wv, Wv16, nw4);
    cvt_kernel<<<(nw4 + 255) / 256, 256, 0, stream>>>(Wo, Wo16, nw4);
    int nr0 = COUTC * DIM * 3, nr1 = COUTC * DIM * 5;
    wrepack_kernel<<<(nr0 + 255) / 256, 256, 0, stream>>>(cw0, wt0, 3);
    wrepack_kernel<<<(nr1 + 255) / 256, 256, 0, stream>>>(cw1, wt1, 5);

    // 1) projections (all-f16 WMMA GEMMs)
    dim3 blk(32);
    dim3 gproj(NTOK / 32, DIM / 64);
    proj_kernel<<<gproj, blk, 0, stream>>>(q16, Wq16, qproj, DIM, DIM, 0);
    proj_kernel<<<gproj, blk, 0, stream>>>(k16, Wk16, kproj, DIM, DIM, 0);
    proj_kernel<<<gproj, blk, 0, stream>>>(v16, Wv16, vT, DIM, DIM, 1);

    // 2) conv branches (implicit GEMM, contiguous f16 weights)
    dim3 gconv(NTOK / 64, COUTC / 16);
    conv_kernel<<<gconv, blk, 0, stream>>>(k16, wt0, cb0, kconv, 3, 0);
    conv_kernel<<<gconv, blk, 0, stream>>>(k16, wt1, cb1, kconv, 5, COUTC);

    // 3) gate blend: kblend = (1-g)*kproj + g*kconv  (folds two score GEMMs into one)
    int nel = NTOK * DIM;
    blend_kernel<<<(nel + 255) / 256, 256, 0, stream>>>(kproj, kconv, gate, kblend);

    // 4) flash attention with async LDS staging
    dim3 gattn(SEQ / 16, BATCH * NHEAD);
    attn_kernel<<<gattn, blk, 0, stream>>>(qproj, kblend, vT, mask, attnO);

    // 5) output projection + bias (fp32 out)
    outproj_kernel<<<gproj, blk, 0, stream>>>(attnO, Wo16, bo, (float*)d_out);
}
